// NIHMultiHeadAttention_34522947125611
// MI455X (gfx1250) — matrix operand order repaired and checked by reading, not compiled, not run on hardware
//
#include <hip/hip_runtime.h>
#include <hip/hip_bf16.h>

// ---------------------------------------------------------------------------
// NIH multi-head attention, CDNA5 (gfx1250), wave32, bf16 WMMA + TDM staging
// N=2048, HID=1024, H=16, D=64
// ---------------------------------------------------------------------------

#define NN   2048
#define HID  1024
#define NH   16
#define DD   64
#define LDAS 2052   // padded LDS leading dim for score strip (bank-conflict break)

typedef __attribute__((ext_vector_type(16))) __bf16 bf16x16;
typedef __attribute__((ext_vector_type(8)))  float  f32x8;
typedef __attribute__((ext_vector_type(4)))  unsigned int u32x4;
typedef __attribute__((ext_vector_type(8)))  unsigned int u32x8;

union FragU {
    bf16x16 v;
    uint4   q[2];
};

// A operand: 16x32 tile from row-major [M,K] matrix, leading dim ld.
// Lanes 0-15: row M=lane, K runs {0..7, 16..23}; lanes 16-31: K runs {8..15, 24..31}.
__device__ inline bf16x16 load_a_frag(const __bf16* base, int ld, int row0, int k0, int lane) {
    int r  = lane & 15;
    int hi = lane >> 4;
    const __bf16* p = base + (size_t)(row0 + r) * ld + k0;
    FragU f;
    f.q[0] = *(const uint4*)(p + (hi ? 8  : 0));
    f.q[1] = *(const uint4*)(p + (hi ? 24 : 16));
    return f.v;
}

// B operand: 32x16 (KxN) tile from row-major B^T [N,K] (ld = K dim).
// Lanes 0-15: col N=lane, K runs {0..7, 16..23}; lanes 16-31: same cols, K runs {8..15, 24..31}.
__device__ inline bf16x16 load_b_frag(const __bf16* bt, int ld, int col0, int k0, int lane) {
    int c  = lane & 15;
    int hi = lane >> 4;
    const __bf16* p = bt + (size_t)(col0 + c) * ld + k0;
    FragU f;
    f.q[0] = *(const uint4*)(p + (hi ? 8  : 0));
    f.q[1] = *(const uint4*)(p + (hi ? 24 : 16));
    return f.v;
}

__device__ inline f32x8 wmma_bf16(bf16x16 a, bf16x16 b, f32x8 c) {
    return __builtin_amdgcn_wmma_f32_16x16x32_bf16(
        /*neg_a=*/false, a, /*neg_b=*/false, b,
        /*c_mod=*/(short)0, c, /*reuse_a=*/false, /*reuse_b=*/false);
}

// ---------------------------------------------------------------------------
// f32 -> bf16 conversion (grid-stride)
// ---------------------------------------------------------------------------
__global__ void cvt_f32_bf16(const float* __restrict__ src, __bf16* __restrict__ dst, int n) {
    int i      = blockIdx.x * blockDim.x + threadIdx.x;
    int stride = gridDim.x * blockDim.x;
    for (; i < n; i += stride) dst[i] = (__bf16)src[i];
}

// ---------------------------------------------------------------------------
// Fused QKV projection: [N,HID] x [3072,HID]^T -> Q,K as [H,N,D], V as [H,D,N]
// One wave per 16x16 output tile.  block = 4 waves, grid = (48, 128)
// ---------------------------------------------------------------------------
__global__ __launch_bounds__(128) void qkv_kernel(
    const __bf16* __restrict__ xb, const __bf16* __restrict__ w3,
    const float* __restrict__ qb, const float* __restrict__ kb, const float* __restrict__ vb,
    __bf16* __restrict__ qws, __bf16* __restrict__ kws, __bf16* __restrict__ vtws) {
    int lane = threadIdx.x & 31;
    int wave = threadIdx.x >> 5;
    int n0 = blockIdx.y * 16;
    int o0 = (blockIdx.x * 4 + wave) * 16;   // 0..3071

    f32x8 acc = {0.f,0.f,0.f,0.f,0.f,0.f,0.f,0.f};
    for (int k0 = 0; k0 < HID; k0 += 32) {
        bf16x16 a = load_a_frag(xb, HID, n0, k0, lane);
        bf16x16 b = load_b_frag(w3, HID, o0, k0, lane);
        acc = wmma_bf16(a, b, acc);
    }

    int t   = o0 >> 10;          // 0=Q 1=K 2=V (uniform per wave)
    int oo0 = o0 & 1023;
    int col = lane & 15, hi = lane >> 4;
    int oo  = oo0 + col;
    const float* bias = (t == 0) ? qb : (t == 1) ? kb : vb;
    float bv = bias[oo];
    int h = oo >> 6, d = oo & 63;

    for (int i = 0; i < 8; i++) {
        int n = n0 + i + hi * 8;
        __bf16 v = (__bf16)(acc[i] + bv);
        if (t == 0)      qws[((size_t)h * NN + n) * DD + d]  = v;
        else if (t == 1) kws[((size_t)h * NN + n) * DD + d]  = v;
        else             vtws[((size_t)h * DD + d) * NN + n] = v;   // V transposed
    }
}

// ---------------------------------------------------------------------------
// Attention: one workgroup per (head, 16-row block).  128 threads (4 waves).
//  - TDM stages the 16x2048 f32 history strip into LDS (single HBM read),
//    overlapped with the QK^T WMMA pass.
//  - Score strip [16 x 2048] lives in LDS; softmax from LDS.
//  - attn / new_history streamed with non-temporal stores.
//  - attn (rebuilt bf16 from LDS) x V via WMMA, K split across 4 waves.
// ---------------------------------------------------------------------------
__global__ __launch_bounds__(128) void attn_kernel(
    const __bf16* __restrict__ qws, const __bf16* __restrict__ kws,
    const __bf16* __restrict__ vtws,
    const float* __restrict__ hist, const float* __restrict__ beta_p,
    float* __restrict__ attn_out, float* __restrict__ nh_out,
    __bf16* __restrict__ ao) {

    __shared__ float sc[16 * LDAS];       // ~128 KB score strip
    __shared__ float hs[16 * NN];         // 128 KB history strip (TDM target)
    __shared__ float pred[8][16];
    __shared__ float mx[16];
    __shared__ float invl[16];
    __shared__ float obuf[4][16][DD];     // 16 KB per-wave partial outputs

    int tid  = threadIdx.x;
    int lane = tid & 31, wave = tid >> 5;
    int h  = blockIdx.y;
    int n0 = blockIdx.x * 16;
    int r  = lane & 15, hi = lane >> 4;

    float beta  = beta_p[0];
    const float scale = 0.125f;           // 1/sqrt(64)

    const __bf16* qh  = qws  + (size_t)h * NN * DD;
    const __bf16* kh  = kws  + (size_t)h * NN * DD;
    const __bf16* vth = vtws + (size_t)h * DD * NN;
    const float*  hh  = hist + (size_t)h * NN * NN;
    float* attn_h = attn_out + (size_t)h * NN * NN;
    float* nh_h   = nh_out   + (size_t)h * NN * NN;

    // ---- kick off TDM: history[h, n0:n0+16, 0:2048] (f32) -> hs[] in LDS
    if (wave == 0) {
        unsigned long long ga = (unsigned long long)(hh + (size_t)n0 * NN);
        unsigned lds_off = (unsigned)(size_t)(&hs[0]);
        u32x4 g0;
        g0[0] = 1u;                                   // count=1, user descriptor
        g0[1] = lds_off;                              // lds_addr
        g0[2] = (unsigned)ga;                         // global_addr[31:0]
        g0[3] = (unsigned)((ga >> 32) & 0x1FFFFFFu)   // global_addr[56:32]
              | 0x80000000u;                          // type=2 ("image")
        u32x8 g1;
        g1[0] = 0x00020000u;                          // data_size=2 (4B), no flags
        g1[1] = (unsigned)NN << 16;                   // tensor_dim0[15:0]=2048
        g1[2] = (unsigned)NN << 16;                   // tensor_dim1[15:0]=2048
        g1[3] = (unsigned)NN << 16;                   // tile_dim0=2048
        g1[4] = 16u;                                  // tile_dim1=16, tile_dim2=0
        g1[5] = (unsigned)NN;                         // tensor_dim0_stride[31:0]=2048
        g1[6] = 0u;                                   // stride hi / dim1_stride lo
        g1[7] = 0u;                                   // dim1_stride hi
        asm volatile("tensor_load_to_lds %0, %1" :: "s"(g0), "s"(g1) : "memory");
    }

    // Q fragments for the 16 rows (K = 0..31 and 32..63)
    bf16x16 a0 = load_a_frag(qh, DD, n0, 0,  lane);
    bf16x16 a1 = load_a_frag(qh, DD, n0, 32, lane);

    // ---- pass 1: scores = (Q K^T) * scale into LDS, wave strips of 512 cols
    int c0w = wave * 512;
    for (int ct = 0; ct < 32; ct++) {
        int m0 = c0w + ct * 16;
        bf16x16 b0 = load_b_frag(kh, DD, m0, 0,  lane);
        bf16x16 b1 = load_b_frag(kh, DD, m0, 32, lane);
        f32x8 accv = {0.f,0.f,0.f,0.f,0.f,0.f,0.f,0.f};
        accv = wmma_bf16(a0, b0, accv);
        accv = wmma_bf16(a1, b1, accv);
        int col = m0 + r;
        for (int i = 0; i < 8; i++)
            sc[(i + hi * 8) * LDAS + col] = accv[i] * scale;
    }

    // ---- history strip ready in LDS
    if (wave == 0) __builtin_amdgcn_s_wait_tensorcnt(0);
    __syncthreads();

    // += beta * history from LDS over the wave's strip
    for (int rr = 0; rr < 16; rr++)
        for (int c = lane; c < 512; c += 32) {
            int col = c0w + c;
            sc[rr * LDAS + col] += beta * hs[rr * NN + col];
        }
    __syncthreads();

    // ---- softmax stats: 8 segments x 16 rows
    {
        int row = tid & 15, seg = tid >> 4;
        float m = -3.4e38f;
        for (int c = seg * 256; c < seg * 256 + 256; c++)
            m = fmaxf(m, sc[row * LDAS + c]);
        pred[seg][row] = m;
        __syncthreads();
        if (tid < 16) {
            float mm = pred[0][tid];
            for (int s = 1; s < 8; s++) mm = fmaxf(mm, pred[s][tid]);
            mx[tid] = mm;
        }
        __syncthreads();
        float mm = mx[row];
        float s = 0.f;
        for (int c = seg * 256; c < seg * 256 + 256; c++)
            s += __expf(sc[row * LDAS + c] - mm);
        pred[seg][row] = s;
        __syncthreads();
        if (tid < 16) {
            float ss = 0.f;
            for (int s = 0; s < 8; s++) ss += pred[s][tid];
            invl[tid] = 1.0f / ss;
        }
        __syncthreads();
    }

    // ---- pass 2a: stream attn + new_history (coalesced, non-temporal)
    for (int rr = 0; rr < 16; rr++) {
        float mmr = mx[rr], il = invl[rr];
        size_t rowoff = (size_t)(n0 + rr) * NN;
        for (int c = lane; c < 512; c += 32) {
            int col = c0w + c;
            float a = __expf(sc[rr * LDAS + col] - mmr) * il;
            __builtin_nontemporal_store(a, &attn_h[rowoff + col]);
            __builtin_nontemporal_store(hs[rr * NN + col] + a, &nh_h[rowoff + col]);
        }
    }

    // ---- pass 2b: out16x64 += attn[16,512] @ V[512,64]  (K split per wave)
    f32x8 oacc[4];
    for (int dt = 0; dt < 4; dt++) oacc[dt] = (f32x8){0.f,0.f,0.f,0.f,0.f,0.f,0.f,0.f};
    {
        float mmr = mx[r], il = invl[r];
        for (int kc = 0; kc < 16; kc++) {
            int c0 = c0w + kc * 32;
            // build bf16 A fragment of attn directly from LDS scores
            FragU fa;
            int base1 = c0 + (hi ? 8  : 0);
            int base2 = c0 + (hi ? 24 : 16);
            for (int j = 0; j < 8; j++) {
                fa.v[j]     = (__bf16)(__expf(sc[r * LDAS + base1 + j] - mmr) * il);
                fa.v[j + 8] = (__bf16)(__expf(sc[r * LDAS + base2 + j] - mmr) * il);
            }
            for (int dt = 0; dt < 4; dt++) {
                bf16x16 b = load_b_frag(vth, NN, dt * 16, c0, lane);
                oacc[dt] = wmma_bf16(fa.v, b, oacc[dt]);
            }
        }
    }
    for (int dt = 0; dt < 4; dt++)
        for (int i = 0; i < 8; i++)
            obuf[wave][i + hi * 8][dt * 16 + r] = oacc[dt][i];
    __syncthreads();

    // reduce 4 wave partials, write head-output as bf16 [N, H*D]
    for (int e = tid; e < 16 * DD; e += 128) {
        int row = e >> 6, d = e & 63;
        float s = obuf[0][row][d] + obuf[1][row][d] + obuf[2][row][d] + obuf[3][row][d];
        ao[(size_t)(n0 + row) * (NH * DD) + h * DD + d] = (__bf16)s;
    }
}

// ---------------------------------------------------------------------------
// Output projection: [N,1024] x [64,1024]^T + b -> [N,64] f32
// ---------------------------------------------------------------------------
__global__ __launch_bounds__(128) void oproj_kernel(
    const __bf16* __restrict__ ao, const __bf16* __restrict__ wob,
    const float* __restrict__ bob, float* __restrict__ out) {
    int lane = threadIdx.x & 31, wave = threadIdx.x >> 5;
    int n0 = blockIdx.x * 16;
    int d0 = wave * 16;

    f32x8 acc = {0.f,0.f,0.f,0.f,0.f,0.f,0.f,0.f};
    for (int k0 = 0; k0 < NH * DD; k0 += 32) {
        bf16x16 a = load_a_frag(ao,  NH * DD, n0, k0, lane);
        bf16x16 b = load_b_frag(wob, NH * DD, d0, k0, lane);
        acc = wmma_bf16(a, b, acc);
    }
    int col = lane & 15, hi = lane >> 4;
    int d = d0 + col;
    float bv = bob[d];
    for (int i = 0; i < 8; i++) {
        int n = n0 + i + hi * 8;
        out[(size_t)n * DD + d] = acc[i] + bv;
    }
}

// ---------------------------------------------------------------------------
extern "C" void kernel_launch(void* const* d_in, const int* in_sizes, int n_in,
                              void* d_out, int out_size, void* d_ws, size_t ws_size,
                              hipStream_t stream) {
    const float* x       = (const float*)d_in[0];
    const float* history = (const float*)d_in[1];
    const float* Wq_w    = (const float*)d_in[2];
    const float* Wq_b    = (const float*)d_in[3];
    const float* Wk_w    = (const float*)d_in[4];
    const float* Wk_b    = (const float*)d_in[5];
    const float* Wv_w    = (const float*)d_in[6];
    const float* Wv_b    = (const float*)d_in[7];
    const float* Wo_w    = (const float*)d_in[8];
    const float* Wo_b    = (const float*)d_in[9];
    const float* beta    = (const float*)d_in[10];

    char* ws = (char*)d_ws;
    size_t off = 0;
    __bf16* xb   = (__bf16*)(ws + off); off += (size_t)NN * HID * 2;        // x bf16
    __bf16* w3   = (__bf16*)(ws + off); off += (size_t)3 * HID * HID * 2;   // Wq|Wk|Wv bf16
    __bf16* wob  = (__bf16*)(ws + off); off += (size_t)DD * HID * 2;        // Wo bf16
    __bf16* qws  = (__bf16*)(ws + off); off += (size_t)NH * NN * DD * 2;    // Q [H,N,D]
    __bf16* kws  = (__bf16*)(ws + off); off += (size_t)NH * NN * DD * 2;    // K [H,N,D]
    __bf16* vtws = (__bf16*)(ws + off); off += (size_t)NH * NN * DD * 2;    // V [H,D,N]
    __bf16* ao   = (__bf16*)(ws + off); off += (size_t)NN * NH * DD * 2;    // attnout [N,H*D]
    (void)ws_size; (void)in_sizes; (void)n_in; (void)out_size;

    float* out    = (float*)d_out;                      // [N, D]
    float* attn   = out + (size_t)NN * DD;              // [H, N, N]
    float* nh     = attn + (size_t)NH * NN * NN;        // [H, N, N]

    cvt_f32_bf16<<<512, 256, 0, stream>>>(x,    xb,              NN * HID);
    cvt_f32_bf16<<<512, 256, 0, stream>>>(Wq_w, w3,              HID * HID);
    cvt_f32_bf16<<<512, 256, 0, stream>>>(Wk_w, w3 + HID * HID,  HID * HID);
    cvt_f32_bf16<<<512, 256, 0, stream>>>(Wv_w, w3 + 2 * HID * HID, HID * HID);
    cvt_f32_bf16<<<64,  256, 0, stream>>>(Wo_w, wob,             DD * HID);

    qkv_kernel<<<dim3(48, 128), 128, 0, stream>>>(xb, w3, Wq_b, Wk_b, Wv_b, qws, kws, vtws);
    attn_kernel<<<dim3(128, 16), 128, 0, stream>>>(qws, kws, vtws, history, beta, attn, nh, ao);
    oproj_kernel<<<128, 128, 0, stream>>>(ao, wob, Wo_b, out);
}
